// MoDABlock_4526895530167
// MI455X (gfx1250) — matrix operand order repaired and checked by reading, not compiled, not run on hardware
//
#include <hip/hip_runtime.h>

// ---------------------------------------------------------------------------
// CDNA5 (gfx1250, wave32) implementation of the MoDA block.
// All GEMMs + attention matmuls run on v_wmma_f32_16x16x32_bf16 (fp32 accum).
// ---------------------------------------------------------------------------

typedef __bf16 bf16;
typedef __attribute__((ext_vector_type(16))) __bf16 v16bf;
typedef __attribute__((ext_vector_type(8)))  __bf16 v8bf;
typedef __attribute__((ext_vector_type(8)))  float  v8f;

static constexpr int  Bb   = 2;
static constexpr int  Tt   = 1024;
static constexpr int  Cc   = 1024;
static constexpr int  HQn  = 16;
static constexpr int  HKn  = 4;
static constexpr int  Dd   = 64;
static constexpr int  HIDn = 4096;
static constexpr int  Ll   = 2;
static constexpr int  BT   = Bb * Tt;                 // 2048
static constexpr int  KEYS = Tt + Tt * Ll;            // 3072
static constexpr float ATT_SCALE = 0.125f;            // 64^-0.5
static constexpr float NEGINF    = -3.0e38f;

// ---- WMMA fragment loaders (wave32 layouts from ISA 7.12.2) ----------------
// A-frag (16-bit, 16xK tile): lane l holds row M = l&15,
//   K halves = {8*hi+0..7} then {8*hi+16..23}  (hi = l>>4)
static __device__ __forceinline__ v16bf ldfragA(const bf16* p) {
  union { v16bf v; v8bf h[2]; } u;
  u.h[0] = *reinterpret_cast<const v8bf*>(p);
  u.h[1] = *reinterpret_cast<const v8bf*>(p + 16);
  return u.v;
}
// B-frag (16-bit, Kx16 tile): lane l holds col N = l&15,
//   K halves = {8*hi+0..7} then {8*hi+16..23}  (hi = l>>4)
static __device__ __forceinline__ v16bf ldfragB(const bf16* p) {
  union { v16bf v; v8bf h[2]; } u;
  u.h[0] = *reinterpret_cast<const v8bf*>(p);
  u.h[1] = *reinterpret_cast<const v8bf*>(p + 16);
  return u.v;
}

#define WMMA_BF16(Af, Bf, Cacc) \
  __builtin_amdgcn_wmma_f32_16x16x32_bf16(false, (Af), false, (Bf), (short)0, (Cacc), false, false)

// ---------------------------------------------------------------------------
// Generic GEMM: Out[N][O] = A[N][K] (bf16, row-major) * W[O][K]^T (bf16)
// One wave -> 32x64 tile (2 M-tiles x 4 N-tiles, fp32 accum).
// ---------------------------------------------------------------------------
__global__ void __launch_bounds__(128)
gemm_bf16_kernel(const bf16* __restrict__ A, const bf16* __restrict__ W,
                 float* __restrict__ outF, bf16* __restrict__ outB,
                 int K, int O) {
  const int lane = threadIdx.x;
  const int lo = lane & 15, hi = lane >> 4;
  const int obase = blockIdx.x * 64;
  const int mbase = blockIdx.y * 128 + threadIdx.y * 32;

  const bf16* a0p = A + (size_t)(mbase +      lo) * K + 8 * hi;
  const bf16* a1p = A + (size_t)(mbase + 16 + lo) * K + 8 * hi;
  const bf16* w0p = W + (size_t)(obase +      lo) * K + 8 * hi;
  const bf16* w1p = W + (size_t)(obase + 16 + lo) * K + 8 * hi;
  const bf16* w2p = W + (size_t)(obase + 32 + lo) * K + 8 * hi;
  const bf16* w3p = W + (size_t)(obase + 48 + lo) * K + 8 * hi;

  v8f acc[2][4] = {};
  for (int k = 0; k < K; k += 32) {
    __builtin_prefetch(a0p + k + 256, 0, 0);   // global_prefetch_b8
    __builtin_prefetch(w0p + k + 256, 0, 0);
    v16bf a[2]  = { ldfragA(a0p + k), ldfragA(a1p + k) };
    v16bf bb[4] = { ldfragB(w0p + k), ldfragB(w1p + k),
                    ldfragB(w2p + k), ldfragB(w3p + k) };
#pragma unroll
    for (int i = 0; i < 2; ++i)
#pragma unroll
      for (int j = 0; j < 4; ++j)
        acc[i][j] = WMMA_BF16(a[i], bb[j], acc[i][j]);
  }

#pragma unroll
  for (int i = 0; i < 2; ++i)
#pragma unroll
    for (int j = 0; j < 4; ++j)
#pragma unroll
      for (int r = 0; r < 8; ++r) {
        const int row = mbase + 16 * i + 8 * hi + r;
        const int col = obase + 16 * j + lo;
        const size_t idx = (size_t)row * O + col;
        if (outF) outF[idx] = acc[i][j][r];
        else      outB[idx] = (bf16)acc[i][j][r];
      }
}

// ---------------------------------------------------------------------------
// Flash-style attention, 1 wave per (b, h, 16-query tile).
// scores^T = K * Q^T keeps the softmax axis on M, so per-query reductions
// need only one shfl_xor(16); exp(scores^T) C-layout == A-layout of P for
// the y = P*V WMMA (zero data movement between the two matmuls).
// ---------------------------------------------------------------------------
static __device__ __forceinline__ void attn_block(
    const bf16* __restrict__ kptr, const bf16* __restrict__ vtptr,
    int absrow, int jbase, int limit, int lo, int hi,
    v16bf qf0, v16bf qf1, float& m_run, float& l_run, v8f (&acc)[4]) {
  v16bf kf00 = ldfragA(kptr + (size_t)(absrow +      lo) * Dd +      8 * hi);
  v16bf kf01 = ldfragA(kptr + (size_t)(absrow +      lo) * Dd + 32 + 8 * hi);
  v16bf kf10 = ldfragA(kptr + (size_t)(absrow + 16 + lo) * Dd +      8 * hi);
  v16bf kf11 = ldfragA(kptr + (size_t)(absrow + 16 + lo) * Dd + 32 + 8 * hi);
  v8f z = {};
  v8f s0 = WMMA_BF16(kf00, qf0, z);  s0 = WMMA_BF16(kf01, qf1, s0);
  v8f s1 = WMMA_BF16(kf10, qf0, z);  s1 = WMMA_BF16(kf11, qf1, s1);

  float p0[8], p1[8];
  float bm = NEGINF;
#pragma unroll
  for (int r = 0; r < 8; ++r) {
    const int j0 = jbase + 8 * hi + r;      // key row of s0[r]
    const int j1 = j0 + 16;                 // key row of s1[r]
    const float a = (j0 <= limit) ? s0[r] * ATT_SCALE : NEGINF;
    const float c = (j1 <= limit) ? s1[r] * ATT_SCALE : NEGINF;
    p0[r] = a; p1[r] = c;
    bm = fmaxf(bm, fmaxf(a, c));
  }
  bm = fmaxf(bm, __shfl_xor(bm, 16));
  const float mnew  = fmaxf(m_run, bm);
  const float alpha = __expf(m_run - mnew);
  float rs = 0.f;
  v16bf pf;
#pragma unroll
  for (int r = 0; r < 8; ++r) {
    const float e0 = __expf(p0[r] - mnew);
    const float e1 = __expf(p1[r] - mnew);
    rs += e0 + e1;
    pf[r]     = (bf16)e0;   // A-layout half j<8  -> key 8*hi + j
    pf[r + 8] = (bf16)e1;   // A-layout half j>=8 -> key 8*hi + 16 + (j-8)
  }
  rs += __shfl_xor(rs, 16);
  l_run = l_run * alpha + rs;
  m_run = mnew;

  // rescale running accumulators: alpha is per-query; acc row q = r + 8*hi
#pragma unroll
  for (int r = 0; r < 8; ++r) {
    const float av = __shfl(alpha, r + 8 * hi);
#pragma unroll
    for (int t = 0; t < 4; ++t) acc[t][r] *= av;
  }
  // y(16q x 64d) += P(16x32) * V(32x64); B-frags from V^T are contiguous
#pragma unroll
  for (int t = 0; t < 4; ++t) {
    v16bf vf = ldfragB(vtptr + (size_t)(16 * t + lo) * KEYS + absrow + 8 * hi);
    acc[t] = WMMA_BF16(pf, vf, acc[t]);
  }
}

__global__ void __launch_bounds__(32)
attn_kernel(const bf16* __restrict__ q_bf, const bf16* __restrict__ k_all,
            const bf16* __restrict__ vt_all, bf16* __restrict__ y_bf) {
  const int lane = threadIdx.x;
  const int lo = lane & 15, hi = lane >> 4;
  const int id = blockIdx.x;
  const int qt = id % (Tt / 16);
  const int h  = (id / (Tt / 16)) % HQn;
  const int b  = id / ((Tt / 16) * HQn);
  const int qbase = qt * 16;
  const int hk = h / (HQn / HKn);

  const bf16* qptr  = q_bf  + (size_t)(b * HQn + h)  * Tt * Dd;
  const bf16* kptr  = k_all + (size_t)(b * HKn + hk) * KEYS * Dd;
  const bf16* vtptr = vt_all+ (size_t)(b * HKn + hk) * Dd * KEYS;

  const v16bf qf0 = ldfragB(qptr + (size_t)(qbase + lo) * Dd +      8 * hi);
  const v16bf qf1 = ldfragB(qptr + (size_t)(qbase + lo) * Dd + 32 + 8 * hi);

  float m_run = NEGINF, l_run = 0.f;
  v8f acc[4] = {};
  const int qtok = qbase + lo;

  const int nA = (qbase + 16 + 31) & ~31;   // self keys (causal), pad to 32
  for (int kb = 0; kb < nA; kb += 32)
    attn_block(kptr, vtptr, kb, kb, qtok, lo, hi, qf0, qf1, m_run, l_run, acc);
  const int nB = 2 * qbase + 32;            // depth keys: j <= 2*qtok+1
  for (int kb = 0; kb < nB; kb += 32)
    attn_block(kptr, vtptr, Tt + kb, kb, 2 * qtok + 1, lo, hi, qf0, qf1,
               m_run, l_run, acc);

  const float linv = 1.f / l_run;
#pragma unroll
  for (int r = 0; r < 8; ++r) {
    const float lv = __shfl(linv, r + 8 * hi);
    const int token = qbase + r + 8 * hi;
    bf16* yp = y_bf + (size_t)(b * Tt + token) * Cc + h * Dd + lo;
#pragma unroll
    for (int t = 0; t < 4; ++t)
      yp[16 * t] = (bf16)(acc[t][r] * lv);
  }
}

// ---------------------------------------------------------------------------
// Elementwise / reshape helpers
// ---------------------------------------------------------------------------
__global__ void __launch_bounds__(256)
rms_kernel(const float* __restrict__ x, const float* __restrict__ res,
           const float* __restrict__ w, bf16* __restrict__ out_bf,
           float* __restrict__ out_sum) {
  __shared__ float red[256];
  const int row = blockIdx.x, tid = threadIdx.x;
  const float* xr = x + (size_t)row * Cc;
  const float* rr = res ? res + (size_t)row * Cc : nullptr;
  float v[4], ss = 0.f;
#pragma unroll
  for (int k = 0; k < 4; ++k) {
    const int c = tid + 256 * k;
    float t = xr[c];
    if (rr) t += rr[c];
    v[k] = t; ss += t * t;
  }
  red[tid] = ss; __syncthreads();
  for (int s = 128; s > 0; s >>= 1) {
    if (tid < s) red[tid] += red[tid + s];
    __syncthreads();
  }
  const float r = rsqrtf(red[0] * (1.f / Cc) + 1e-5f);
#pragma unroll
  for (int k = 0; k < 4; ++k) {
    const int c = tid + 256 * k;
    if (out_sum) out_sum[(size_t)row * Cc + c] = v[k];
    out_bf[(size_t)row * Cc + c] = (bf16)(v[k] * r * w[c]);
  }
}

__global__ void __launch_bounds__(256)
cvt_kernel(const float* __restrict__ in, bf16* __restrict__ out, int n) {
  const int i = blockIdx.x * 256 + threadIdx.x;
  if (i < n) out[i] = (bf16)in[i];
}

__global__ void __launch_bounds__(256)
rope_kernel(const float* __restrict__ in, const float* __restrict__ cosT,
            const float* __restrict__ sinT, bf16* __restrict__ out,
            int H, int ROWS) {
  const int i = blockIdx.x * 256 + threadIdx.x;
  const int total = BT * H * (Dd / 2);
  if (i >= total) return;
  const int p = i & 31;
  const int h = (i >> 5) % H;
  const int t = ((i >> 5) / H) % Tt;
  const int b = i / (32 * H * Tt);
  const float* row = in + (size_t)(b * Tt + t) * (H * Dd) + h * Dd + 2 * p;
  const float a = row[0], bv = row[1];
  const float c = cosT[t * 32 + p], s = sinT[t * 32 + p];
  const size_t o = ((size_t)(b * H + h) * ROWS + t) * Dd + 2 * p;
  out[o]     = (bf16)(a * c - bv * s);
  out[o + 1] = (bf16)(a * s + bv * c);
}

__global__ void __launch_bounds__(256)
vt_kernel(const float* __restrict__ v, bf16* __restrict__ vt) {
  const int i = blockIdx.x * 256 + threadIdx.x;   // v: [B*T][HK*64]
  if (i >= BT * HKn * Dd) return;
  const int d = i & 63, h = (i >> 6) & 3;
  const int bt = i >> 8, t = bt & 1023, b = bt >> 10;
  vt[((size_t)(b * HKn + h) * Dd + d) * KEYS + t] = (bf16)v[i];
}

__global__ void __launch_bounds__(256)
depth_kernel(const float* __restrict__ dk, const float* __restrict__ dv,
             bf16* __restrict__ k_all, bf16* __restrict__ vt_all) {
  const int i = blockIdx.x * 256 + threadIdx.x;   // (L,B,HK,T,D) flat
  if (i >= Ll * Bb * HKn * Tt * Dd) return;
  const int d = i & 63;
  const int t = (i >> 6) & 1023;
  const int h = (i >> 16) & 3;
  const int b = (i >> 18) & 1;
  const int l = i >> 19;
  const int key = Tt + t * Ll + l;
  k_all [((size_t)(b * HKn + h) * KEYS + key) * Dd + d]  = (bf16)dk[i];
  vt_all[((size_t)(b * HKn + h) * Dd + d) * KEYS + key]  = (bf16)dv[i];
}

__global__ void __launch_bounds__(256)
silu_kernel(bf16* __restrict__ g, const bf16* __restrict__ u) {
  const int i = blockIdx.x * 256 + threadIdx.x;
  if (i >= BT * HIDn) return;
  const float gv = (float)g[i], uv = (float)u[i];
  g[i] = (bf16)((gv / (1.f + __expf(-gv))) * uv);
}

__global__ void __launch_bounds__(256)
addcvt_kernel(const float* __restrict__ a, const float* __restrict__ b,
              float* __restrict__ outF, bf16* __restrict__ outB) {
  const int i = blockIdx.x * 256 + threadIdx.x;
  if (i >= BT * Cc) return;
  const float s = a[i] + b[i];
  outF[i] = s; outB[i] = (bf16)s;
}

__global__ void __launch_bounds__(256)
perm_kernel(const float* __restrict__ in, float* __restrict__ out) {
  const int i = blockIdx.x * 256 + threadIdx.x;   // [B*T][HK*64] -> [B,HK,T,D]
  if (i >= BT * HKn * Dd) return;
  const int d = i & 63, h = (i >> 6) & 3;
  const int bt = i >> 8, t = bt & 1023, b = bt >> 10;
  out[((size_t)(b * HKn + h) * Tt + t) * Dd + d] = in[i];
}

// ---------------------------------------------------------------------------
extern "C" void kernel_launch(void* const* d_in, const int* in_sizes, int n_in,
                              void* d_out, int out_size, void* d_ws, size_t ws_size,
                              hipStream_t stream) {
  (void)in_sizes; (void)n_in; (void)out_size; (void)ws_size;
  const float* x_in   = (const float*)d_in[0];
  const float* cosT   = (const float*)d_in[1];
  const float* sinT   = (const float*)d_in[2];
  const float* dkeys  = (const float*)d_in[3];
  const float* dvals  = (const float*)d_in[4];
  const float* ln1    = (const float*)d_in[5];
  const float* ln2    = (const float*)d_in[6];
  const float* Wsrc[11] = {
    (const float*)d_in[7],  (const float*)d_in[8],  (const float*)d_in[9],
    (const float*)d_in[10], (const float*)d_in[11], (const float*)d_in[12],
    (const float*)d_in[13], (const float*)d_in[14], (const float*)d_in[15],
    (const float*)d_in[16], (const float*)d_in[17] };
  const int Wnum[11] = { Cc*Cc, 256*Cc, 256*Cc, Cc*Cc, 256*Cc, 256*Cc,
                         HIDn*Cc, HIDn*Cc, Cc*HIDn, 256*Cc, 256*Cc };

  // ---- workspace layout (~104 MB) ----
  size_t off = 0;
  auto alloc = [&](size_t bytes) -> void* {
    void* p = (char*)d_ws + off;
    off += (bytes + 255) & ~(size_t)255;
    return p;
  };
  bf16* h_bf   = (bf16*)alloc((size_t)BT * Cc * 2);          // h / h2 (reused)
  bf16* Wbf[11];
  for (int i = 0; i < 11; ++i) Wbf[i] = (bf16*)alloc((size_t)Wnum[i] * 2);
  bf16* q_bf   = (bf16*)alloc((size_t)BT * Cc * 2);          // q / x2_bf (reused)
  bf16* k_all  = (bf16*)alloc((size_t)Bb * HKn * KEYS * Dd * 2);
  bf16* vt_all = (bf16*)alloc((size_t)Bb * HKn * Dd * KEYS * 2);
  bf16* y_bf   = (bf16*)alloc((size_t)BT * Cc * 2);
  float* big   = (float*)alloc((size_t)BT * Cc * 4);         // q/Wo/Wd outputs
  float* p256  = (float*)alloc((size_t)BT * 256 * 4);        // k/v/kd/vd/kf/vf
  float* x1    = (float*)alloc((size_t)BT * Cc * 4);
  bf16* g_bf   = (bf16*)alloc((size_t)BT * HIDn * 2);
  bf16* u_bf   = (bf16*)alloc((size_t)BT * HIDn * 2);

  float* out_x   = (float*)d_out;
  float* out_kwa = out_x + 2097152;
  float* out_kwf = out_x + 2621440;
  float* out_vwa = out_x + 3145728;
  float* out_vwf = out_x + 3670016;

  auto gemm = [&](const bf16* A, const bf16* W, float* oF, bf16* oB, int O, int K) {
    gemm_bf16_kernel<<<dim3(O / 64, BT / 128), dim3(32, 4), 0, stream>>>(A, W, oF, oB, K, O);
  };

  // 1) h = rms(x, ln1) -> bf16
  rms_kernel<<<BT, 256, 0, stream>>>(x_in, nullptr, ln1, h_bf, nullptr);
  // 2) weights -> bf16
  for (int i = 0; i < 11; ++i)
    cvt_kernel<<<(Wnum[i] + 255) / 256, 256, 0, stream>>>(Wsrc[i], Wbf[i], Wnum[i]);
  // 3) q proj + rope
  gemm(h_bf, Wbf[0], big, nullptr, Cc, Cc);
  rope_kernel<<<(BT*HQn*32 + 255)/256, 256, 0, stream>>>(big, cosT, sinT, q_bf, HQn, Tt);
  // 4) k proj + rope -> k_all[:, :1024]
  gemm(h_bf, Wbf[1], p256, nullptr, 256, Cc);
  rope_kernel<<<(BT*HKn*32 + 255)/256, 256, 0, stream>>>(p256, cosT, sinT, k_all, HKn, KEYS);
  // 5) v proj -> vt_all[:, :, :1024] (transposed)
  gemm(h_bf, Wbf[2], p256, nullptr, 256, Cc);
  vt_kernel<<<(BT*HKn*Dd + 255)/256, 256, 0, stream>>>(p256, vt_all);
  // 6) depth K/V -> k_all / vt_all tails
  depth_kernel<<<(Ll*Bb*HKn*Tt*Dd + 255)/256, 256, 0, stream>>>(dkeys, dvals, k_all, vt_all);
  // 7) attention write heads (use h): Wdk, Wdv
  gemm(h_bf, Wbf[4], p256, nullptr, 256, Cc);
  perm_kernel<<<(BT*HKn*Dd + 255)/256, 256, 0, stream>>>(p256, out_kwa);
  gemm(h_bf, Wbf[5], p256, nullptr, 256, Cc);
  perm_kernel<<<(BT*HKn*Dd + 255)/256, 256, 0, stream>>>(p256, out_vwa);
  // 8) attention
  attn_kernel<<<Bb * HQn * (Tt / 16), 32, 0, stream>>>(q_bf, k_all, vt_all, y_bf);
  // 9) o = y @ Wo^T ; x1 = x + o ; h2 = rms(x1, ln2)
  gemm(y_bf, Wbf[3], big, nullptr, Cc, Cc);
  rms_kernel<<<BT, 256, 0, stream>>>(x_in, big, ln2, h_bf, x1);
  // 10) MLP
  gemm(h_bf, Wbf[6], nullptr, g_bf, HIDn, Cc);
  gemm(h_bf, Wbf[7], nullptr, u_bf, HIDn, Cc);
  silu_kernel<<<(BT*HIDn + 255)/256, 256, 0, stream>>>(g_bf, u_bf);
  gemm(g_bf, Wbf[8], big, nullptr, Cc, HIDn);
  // 11) x = x1 + mlp  (fp32 output + bf16 copy for final projections)
  addcvt_kernel<<<(BT*Cc + 255)/256, 256, 0, stream>>>(x1, big, out_x, q_bf);
  // 12) ffn write heads (use final x): Wfk, Wfv
  gemm(q_bf, Wbf[9], p256, nullptr, 256, Cc);
  perm_kernel<<<(BT*HKn*Dd + 255)/256, 256, 0, stream>>>(p256, out_kwf);
  gemm(q_bf, Wbf[10], p256, nullptr, 256, Cc);
  perm_kernel<<<(BT*HKn*Dd + 255)/256, 256, 0, stream>>>(p256, out_vwf);
}
